// PAM_3289944949056
// MI455X (gfx1250) — matrix operand order repaired and checked by reading, not compiled, not run on hardware
//
#include <hip/hip_runtime.h>
#include <hip/hip_bf16.h>

typedef __attribute__((ext_vector_type(16))) __bf16 v16bf;
typedef __attribute__((ext_vector_type(8)))  float  v8f;
typedef __attribute__((ext_vector_type(4)))  int    v4i;

typedef __attribute__((address_space(1))) v4i* as1_v4i_p;
typedef __attribute__((address_space(3))) v4i* as3_v4i_p;

#define C_DIM  256
#define NPIX   4096
#define NBATCH 8

// Build a v16bf from two 16-byte chunks (32B WMMA operands split into 2xb128).
__device__ __forceinline__ v16bf ld2x16(const void* lo, const void* hi) {
    union { v16bf v; uint4 u[2]; } t;
    t.u[0] = *(const uint4*)lo;
    t.u[1] = *(const uint4*)hi;
    return t.v;
}

__device__ __forceinline__ v8f wmma_bf16(v16bf a, v16bf b, v8f c) {
    return __builtin_amdgcn_wmma_f32_16x16x32_bf16(
        false, a, false, b, (short)0, c, false, false);
}

// --- CDNA5 async global->LDS copy (ASYNCcnt-tracked), with sync fallback ----
__device__ __forceinline__ void async_ld16(const void* g, void* l) {
#if __has_builtin(__builtin_amdgcn_global_load_async_to_lds_b128)
    __builtin_amdgcn_global_load_async_to_lds_b128(
        (as1_v4i_p)(g), (as3_v4i_p)(l), 0, 0);
#else
    *(uint4*)l = *(const uint4*)g;
#endif
}

template <int N>
__device__ __forceinline__ void wait_async() {
#if __has_builtin(__builtin_amdgcn_s_wait_asynccnt)
    __builtin_amdgcn_s_wait_asynccnt(N);
#else
    asm volatile("s_wait_asynccnt %0" :: "n"(N) : "memory");
#endif
}

// ---------------------------------------------------------------------------
// Projection kernel: out[o,l] = sum_c W[o,c] x[c,l] + bias[o]  (per batch)
// grid = (64 pixel-tiles, 8 batches, 3 projections), block = 256 (8 waves)
// proj 0 -> Bq [n][pix][ch], proj 1 -> Ct [n][pix][ch], proj 2 -> Dt [n][ch][pix]
// ---------------------------------------------------------------------------
__global__ __launch_bounds__(256)
void pam_proj_kernel(const float* __restrict__ x,
                     const float* __restrict__ Wb, const float* __restrict__ bb,
                     const float* __restrict__ Wc, const float* __restrict__ bc,
                     const float* __restrict__ Wd, const float* __restrict__ bd,
                     unsigned short* __restrict__ ws_raw)
{
    __bf16* ws = (__bf16*)ws_raw;
    __bf16* bqd = ws;
    __bf16* ctd = ws + (size_t)NBATCH * NPIX * C_DIM;
    __bf16* dtd = ws + (size_t)2 * NBATCH * NPIX * C_DIM;

    const int ptile = blockIdx.x;
    const int b     = blockIdx.y;
    const int proj  = blockIdx.z;

    const float* W;  const float* bias; __bf16* dst; bool chanMajor;
    if (proj == 0)      { W = Wb; bias = bb; dst = bqd; chanMajor = false; }
    else if (proj == 1) { W = Wc; bias = bc; dst = ctd; chanMajor = false; }
    else                { W = Wd; bias = bd; dst = dtd; chanMajor = true;  }

    const int tid = threadIdx.x;
    const int p0  = ptile * 64;

    __shared__ __align__(32) __bf16 xs[64][C_DIM + 8];
    {
        const float* xb = x + (size_t)b * C_DIM * NPIX;
        for (int idx = tid; idx < C_DIM * 64; idx += 256) {
            int ch = idx >> 6;
            int p  = idx & 63;
            xs[p][ch] = (__bf16)xb[(size_t)ch * NPIX + p0 + p];
        }
    }
    __syncthreads();

    const int wv   = tid >> 5;
    const int lane = tid & 31;
    const int lh   = lane >> 4;
    const int ln   = lane & 15;
    const int obase = wv * 32;

    v8f acc[2][4];
    #pragma unroll
    for (int ot = 0; ot < 2; ++ot)
        #pragma unroll
        for (int pt = 0; pt < 4; ++pt)
            #pragma unroll
            for (int e = 0; e < 8; ++e) acc[ot][pt][e] = 0.0f;

    for (int c0 = 0; c0 < C_DIM; c0 += 32) {
        v16bf a[2];
        #pragma unroll
        for (int ot = 0; ot < 2; ++ot) {
            int o = obase + 16 * ot + ln;
            const float* wrow = W + (size_t)o * C_DIM + c0 + lh * 8;
            v16bf av;
            #pragma unroll
            for (int e = 0; e < 8; ++e) av[e]     = (__bf16)wrow[e];
            #pragma unroll
            for (int e = 0; e < 8; ++e) av[8 + e] = (__bf16)wrow[16 + e];
            a[ot] = av;
        }
        #pragma unroll
        for (int pt = 0; pt < 4; ++pt) {
            const __bf16* xr = &xs[pt * 16 + ln][c0 + lh * 8];
            v16bf bv = ld2x16(xr, xr + 16);
            #pragma unroll
            for (int ot = 0; ot < 2; ++ot)
                acc[ot][pt] = wmma_bf16(a[ot], bv, acc[ot][pt]);
        }
    }

    #pragma unroll
    for (int ot = 0; ot < 2; ++ot) {
        const int orow0 = obase + 16 * ot + lh * 8;
        float bv[8];
        #pragma unroll
        for (int r = 0; r < 8; ++r) bv[r] = bias[orow0 + r];
        #pragma unroll
        for (int pt = 0; pt < 4; ++pt) {
            const int p = p0 + pt * 16 + ln;
            #pragma unroll
            for (int r = 0; r < 8; ++r) {
                float v = acc[ot][pt][r] + bv[r];
                int o = orow0 + r;
                size_t idx = chanMajor
                    ? ((size_t)(b * C_DIM + o)) * NPIX + p
                    : ((size_t)(b * NPIX + p)) * C_DIM + o;
                dst[idx] = (__bf16)v;
            }
        }
    }
}

// ---------------------------------------------------------------------------
// Flash-attention kernel: out = alpha * softmax(B C) D + x
// grid = (64 query-tiles of 64, 8 batches), block = 128 (4 waves).
// K/V tiles staged once per workgroup via async global->LDS, double buffered.
// ---------------------------------------------------------------------------
__global__ __launch_bounds__(128)
void pam_flash_kernel(const float* __restrict__ x,
                      const float* __restrict__ alpha_p,
                      const unsigned short* __restrict__ ws_raw,
                      float* __restrict__ out)
{
    const __bf16* ws  = (const __bf16*)ws_raw;
    const __bf16* bqd = ws;
    const __bf16* ctd = ws + (size_t)NBATCH * NPIX * C_DIM;
    const __bf16* dtd = ws + (size_t)2 * NBATCH * NPIX * C_DIM;

    const int b     = blockIdx.y;
    const int q0blk = blockIdx.x * 64;
    const int tid   = threadIdx.x;
    const int wv    = tid >> 5;
    const int lane  = tid & 31;
    const int lh    = lane >> 4;
    const int ln    = lane & 15;

    __shared__ __align__(32) __bf16 qs[64][C_DIM + 8];   // Q block
    __shared__ __align__(32) __bf16 ps[4][16][40];       // P restage per wave
    __shared__ __align__(32) __bf16 ks[2][32][C_DIM + 8];// K tiles (dbl buf)
    __shared__ __align__(32) __bf16 vsm[2][C_DIM][40];   // V tiles (dbl buf)

    const __bf16* ctb = ctd + (size_t)b * NPIX * C_DIM;
    const __bf16* dtb = dtd + (size_t)b * C_DIM * NPIX;

    // Cooperative Q load (coalesced 16B chunks).
    {
        const uint4* src = (const uint4*)(bqd + ((size_t)(b * NPIX + q0blk)) * C_DIM);
        for (int i = tid; i < 64 * (C_DIM / 8); i += 128) {
            int row = i >> 5;
            int col = i & 31;
            ((uint4*)&qs[row][0])[col] = src[row * 32 + col];
        }
    }

    // Issue async staging of a 32-key K/V tile (16 x b128 per thread).
    auto stage_tiles = [&](int j0, int buf) {
        #pragma unroll
        for (int i = 0; i < 8; ++i) {           // K: 32 keys x 512B
            int idx = tid + 128 * i;
            int key = idx >> 5, chunk = idx & 31;
            async_ld16(ctb + ((size_t)(j0 + key)) * C_DIM + chunk * 8,
                       &ks[buf][key][chunk * 8]);
        }
        #pragma unroll
        for (int i = 0; i < 8; ++i) {           // V: 256 ch x 64B
            int idx = tid + 128 * i;
            int ch = idx >> 2, chunk = idx & 3;
            async_ld16(dtb + ((size_t)ch) * NPIX + j0 + chunk * 8,
                       &vsm[buf][ch][chunk * 8]);
        }
    };

    const int qs_row = wv * 16 + ln;

    v8f Y[16];
    #pragma unroll
    for (int t = 0; t < 16; ++t)
        #pragma unroll
        for (int e = 0; e < 8; ++e) Y[t][e] = 0.0f;
    float m[8], l[8];
    #pragma unroll
    for (int r = 0; r < 8; ++r) { m[r] = -1e30f; l[r] = 0.0f; }

    stage_tiles(0, 0);

    for (int jt = 0; jt < NPIX / 32; ++jt) {
        const int buf = jt & 1;
        if (jt + 1 < NPIX / 32) {
            stage_tiles((jt + 1) * 32, buf ^ 1);
            wait_async<16>();   // in-order: tile jt's 16 loads have landed
        } else {
            wait_async<0>();
        }
        __syncthreads();        // tile visible to all 4 waves (incl. Q on jt==0)

        // ---- scores S = Q . K^T for 32 keys (from LDS) ---------------------
        v8f s0, s1;
        #pragma unroll
        for (int e = 0; e < 8; ++e) { s0[e] = 0.0f; s1[e] = 0.0f; }
        #pragma unroll
        for (int c0 = 0; c0 < C_DIM; c0 += 32) {
            const __bf16* qr = &qs[qs_row][c0 + lh * 16];
            v16bf qa = ld2x16(qr, qr + 8);
            const __bf16* k0p = &ks[buf][ln][c0 + lh * 16];
            const __bf16* k1p = &ks[buf][16 + ln][c0 + lh * 16];
            v16bf kb0 = ld2x16(k0p, k0p + 8);
            v16bf kb1 = ld2x16(k1p, k1p + 8);
            s0 = wmma_bf16(qa, kb0, s0);
            s1 = wmma_bf16(qa, kb1, s1);
        }

        // ---- online softmax: rows live across 16-lane halves ---------------
        float p0[8], p1[8];
        #pragma unroll
        for (int r = 0; r < 8; ++r) {
            float t = fmaxf(s0[r], s1[r]);
            t = fmaxf(t, __shfl_xor(t, 1, 32));
            t = fmaxf(t, __shfl_xor(t, 2, 32));
            t = fmaxf(t, __shfl_xor(t, 4, 32));
            t = fmaxf(t, __shfl_xor(t, 8, 32));
            float mn = fmaxf(m[r], t);
            float sc = __expf(m[r] - mn);
            m[r] = mn;
            float e0 = __expf(s0[r] - mn);
            float e1 = __expf(s1[r] - mn);
            p0[r] = e0; p1[r] = e1;
            float rs = e0 + e1;
            rs += __shfl_xor(rs, 1, 32);
            rs += __shfl_xor(rs, 2, 32);
            rs += __shfl_xor(rs, 4, 32);
            rs += __shfl_xor(rs, 8, 32);
            l[r] = l[r] * sc + rs;
            #pragma unroll
            for (int t2 = 0; t2 < 16; ++t2) Y[t2][r] *= sc;
        }

        // ---- restage P: C/D layout -> A layout via LDS ---------------------
        #pragma unroll
        for (int r = 0; r < 8; ++r) {
            int row = r + lh * 8;
            ps[wv][row][ln]      = (__bf16)p0[r];
            ps[wv][row][16 + ln] = (__bf16)p1[r];
        }
        asm volatile("s_wait_dscnt 0" ::: "memory");  // wave-local LDS ordering

        const __bf16* pr = &ps[wv][ln][lh * 16];
        v16bf pa = ld2x16(pr, pr + 8);

        // ---- Y += P . V over all 16 channel tiles (V from LDS) -------------
        #pragma unroll
        for (int t = 0; t < 16; ++t) {
            const __bf16* vp = &vsm[buf][t * 16 + ln][lh * 16];
            v16bf vb = ld2x16(vp, vp + 8);
            Y[t] = wmma_bf16(pa, vb, Y[t]);
        }

        __syncthreads();        // before buf is overwritten by async writes
    }

    // ---- epilogue: out = alpha * (Y / l) + x -------------------------------
    const float alpha = *alpha_p;
    const int qbase = q0blk + wv * 16 + lh * 8;
    #pragma unroll
    for (int r = 0; r < 8; ++r) {
        float inv = 1.0f / l[r];
        int q = qbase + r;
        #pragma unroll
        for (int t = 0; t < 16; ++t) {
            int ch = t * 16 + ln;
            size_t idx = ((size_t)(b * C_DIM + ch)) * NPIX + q;
            out[idx] = alpha * (Y[t][r] * inv) + x[idx];
        }
    }
}

extern "C" void kernel_launch(void* const* d_in, const int* in_sizes, int n_in,
                              void* d_out, int out_size, void* d_ws, size_t ws_size,
                              hipStream_t stream) {
    const float* x     = (const float*)d_in[0];
    const float* Wb    = (const float*)d_in[1];
    const float* bb    = (const float*)d_in[2];
    const float* Wc    = (const float*)d_in[3];
    const float* bc    = (const float*)d_in[4];
    const float* Wd    = (const float*)d_in[5];
    const float* bd    = (const float*)d_in[6];
    const float* alpha = (const float*)d_in[7];
    float* out = (float*)d_out;
    unsigned short* ws = (unsigned short*)d_ws;

    dim3 gp(NPIX / 64, NBATCH, 3);
    pam_proj_kernel<<<gp, 256, 0, stream>>>(x, Wb, bb, Wc, bc, Wd, bd, ws);

    dim3 gf(NPIX / 64, NBATCH);
    pam_flash_kernel<<<gf, 128, 0, stream>>>(x, alpha, ws, out);
}
